// Block_69861938036758
// MI455X (gfx1250) — matrix operand order repaired and checked by reading, not compiled, not run on hardware
//
#include <hip/hip_runtime.h>
#include <math.h>

typedef _Float16 f16;
typedef __attribute__((ext_vector_type(16))) _Float16 v16h;
typedef __attribute__((ext_vector_type(8)))  _Float16 v8h;
typedef __attribute__((ext_vector_type(8)))  float    v8f;

#define DMODEL 768
#define NSEQ   1024
#define BATCH  8
#define NHEAD  12
#define HDIM   64
#define BHD    (BATCH*NHEAD)
#define ROWS   (BATCH*NSEQ)
#define ATT_SCALE 0.125f
#define NEGINF (-__builtin_inff())

// ---------------- WMMA fragment helpers (gfx1250 wave32) ----------------
static __device__ __forceinline__ v16h cat8(v8h a, v8h b){
  return __builtin_shufflevector(a,b,0,1,2,3,4,5,6,7,8,9,10,11,12,13,14,15);
}
// A-matrix 16x32 f16: lane m=lane&15; lo half-wave K 0-15, hi half K 16-31
static __device__ __forceinline__ v16h load_frag_a(const f16* __restrict__ p, int ld, int lane){
  int m = lane & 15, hi = lane >> 4;
  const f16* r = p + m*ld + hi*16;
  return cat8(*(const v8h*)r, *(const v8h*)(r+8));
}
// B-matrix 32x16 f16 from B^T row-major: lane n holds column n; lanes 0-15: K 0-15, lanes 16-31: K 16-31
static __device__ __forceinline__ v16h load_frag_b(const f16* __restrict__ p, int ld, int lane){
  int n = lane & 15, hi = lane >> 4;
  const f16* r = p + n*ld + hi*16;
  return cat8(*(const v8h*)r, *(const v8h*)(r+8));
}
static __device__ __forceinline__ v8f wmma32(v16h a, v16h b, v8f c){
  return __builtin_amdgcn_wmma_f32_16x16x32_f16(false, a, false, b, (short)0, c, false, false);
}
static __device__ __forceinline__ unsigned pack2h(float a, float b){
  union { f16 h[2]; unsigned u; } x;
  x.h[0] = (f16)a; x.h[1] = (f16)b; return x.u;
}

// ---------------- small utility kernels ----------------
__global__ void cvt_f32_f16_kernel(const float* __restrict__ in, f16* __restrict__ out, int n){
  int i = blockIdx.x*blockDim.x + threadIdx.x;
  if (i < n) out[i] = (f16)in[i];
}

__global__ void maskbias_kernel(const int* __restrict__ mask, float* __restrict__ bias){
  int i = blockIdx.x*blockDim.x + threadIdx.x;
  if (i < ROWS) bias[i] = (mask[i] == 0) ? NEGINF : 0.0f;
}

// LayerNorm over 768, one row per block (256 threads), f32 in -> f16 out
__global__ void ln_kernel(const float* __restrict__ x, const float* __restrict__ w,
                          const float* __restrict__ b, f16* __restrict__ out){
  int row = blockIdx.x;
  int t = threadIdx.x;
  const float* xr = x + (size_t)row*DMODEL;
  float v0 = xr[t], v1 = xr[t+256], v2 = xr[t+512];
  float s  = v0+v1+v2;
  float sq = v0*v0+v1*v1+v2*v2;
  for (int off = 16; off > 0; off >>= 1){
    s  += __shfl_xor(s,  off, 32);
    sq += __shfl_xor(sq, off, 32);
  }
  __shared__ float red[16];
  int wid = t >> 5;
  if ((t & 31) == 0){ red[wid] = s; red[8+wid] = sq; }
  __syncthreads();
  float S = 0.f, SQ = 0.f;
  for (int i = 0; i < 8; i++){ S += red[i]; SQ += red[8+i]; }
  float mean = S * (1.0f/DMODEL);
  float var  = SQ * (1.0f/DMODEL) - mean*mean;
  float rstd = rsqrtf(var + 1e-5f);
  f16* o = out + (size_t)row*DMODEL;
  o[t]     = (f16)((v0-mean)*rstd*w[t]     + b[t]);
  o[t+256] = (f16)((v1-mean)*rstd*w[t+256] + b[t+256]);
  o[t+512] = (f16)((v2-mean)*rstd*w[t+512] + b[t+512]);
}

// ---------------- generic WMMA GEMM: C = A(MxK) @ W(NoutxK)^T ----------------
// 64x64 per wave (4x4 WMMA tiles, 16 WMMA per k-step for 8 fragment loads),
// 4 waves/block arranged 2x2 -> 128x128 per block.
// EPI 0: QKV scatter to q[b,h,n,hd], k[b,h,n,hd], v_t[b,h,hd,n] (f16)
// EPI 1: outF = acc + bias[f] + resid (f32)
// EPI 2: outH = gelu(acc + bias[f]) (f16)
template<int EPI>
__global__ void gemm_kernel(const f16* __restrict__ A, const f16* __restrict__ W,
                            int K, int Nout,
                            const float* __restrict__ bias, const float* __restrict__ resid,
                            float* __restrict__ outF, f16* __restrict__ outH,
                            f16* __restrict__ qp, f16* __restrict__ kp, f16* __restrict__ vp,
                            int seg_off){
  int lane = threadIdx.x & 31;
  int wave = threadIdx.x >> 5;      // 4 waves: 2x2 of 64x64 tiles -> 128x128 per block
  int wm = wave & 1, wn = wave >> 1;
  int n0 = blockIdx.x*128 + wn*64;
  int m0 = blockIdx.y*128 + wm*64;  // segment-local (EPI 0) or global row
  int arow = (EPI == 0) ? ((int)blockIdx.z*NSEQ + seg_off + m0) : m0;

  v8f c[4][4] = {};
  for (int k0 = 0; k0 < K; k0 += 32){
    v16h af[4], bf[4];
#pragma unroll
    for (int mi = 0; mi < 4; mi++)
      af[mi] = load_frag_a(A + (size_t)(arow + mi*16)*K + k0, K, lane);
#pragma unroll
    for (int ni = 0; ni < 4; ni++)
      bf[ni] = load_frag_b(W + (size_t)(n0 + ni*16)*K + k0, K, lane);
#pragma unroll
    for (int mi = 0; mi < 4; mi++)
#pragma unroll
      for (int ni = 0; ni < 4; ni++)
        c[mi][ni] = wmma32(af[mi], bf[ni], c[mi][ni]);
  }

  int col = lane & 15, hi = lane >> 4;
#pragma unroll
  for (int mi = 0; mi < 4; mi++)
#pragma unroll
  for (int ni = 0; ni < 4; ni++){
    for (int r = 0; r < 8; r++){
      int mrow = m0 + mi*16 + r + 8*hi;
      int f    = n0 + ni*16 + col;
      float v  = c[mi][ni][r];
      if (EPI == 0){
        int b  = blockIdx.z;
        int n  = seg_off + mrow;
        int cc = f / DMODEL;
        int rem = f - cc*DMODEL;
        int h  = rem >> 6;
        int hd = rem & 63;
        f16 hv = (f16)v;
        if (cc == 0)      qp[(((size_t)b*NHEAD + h)*NSEQ + n)*HDIM + hd] = hv;
        else if (cc == 1) kp[(((size_t)b*NHEAD + h)*NSEQ + n)*HDIM + hd] = hv;
        else              vp[(((size_t)b*NHEAD + h)*HDIM + hd)*NSEQ + n] = hv;
      } else if (EPI == 1){
        size_t idx = (size_t)mrow*Nout + f;
        outF[idx] = v + bias[f] + resid[idx];
      } else {
        float t = v + bias[f];
        float g = 0.5f*t*(1.0f + erff(t*0.70710678118654752f));
        outH[(size_t)mrow*Nout + f] = (f16)g;
      }
    }
  }
}

// ---------------- segment-mean pass A: qsum over all q, masked ksum per segment ----------------
__global__ void sumqk_kernel(const f16* __restrict__ q, const f16* __restrict__ k,
                             const float* __restrict__ mbias,
                             float* __restrict__ qsum, float* __restrict__ ksum){
  int bh = blockIdx.x;          // 96
  int b  = bh / NHEAD;
  int t  = threadIdx.x;         // 256
  int d  = t & 63, g = t >> 6;  // 4 groups
  const f16* qb = q + (size_t)bh*NSEQ*HDIM;
  const f16* kb = k + (size_t)bh*NSEQ*HDIM;
  const float* mb = mbias + b*NSEQ;
  float qs = 0.f, ks0 = 0.f, ks1 = 0.f, ks2 = 0.f;
  for (int n = g; n < NSEQ; n += 4){
    qs += (float)qb[n*HDIM + d];
    if (mb[n] == 0.0f){
      float kv = (float)kb[n*HDIM + d];
      if (n < 256)      ks0 += kv;
      else if (n < 768) ks1 += kv;
      else              ks2 += kv;
    }
  }
  __shared__ float red[4][4][64];
  red[g][0][d] = qs; red[g][1][d] = ks0; red[g][2][d] = ks1; red[g][3][d] = ks2;
  __syncthreads();
  if (g == 0){
    float a0=0,a1=0,a2=0,a3=0;
    for (int i = 0; i < 4; i++){
      a0 += red[i][0][d]; a1 += red[i][1][d]; a2 += red[i][2][d]; a3 += red[i][3][d];
    }
    qsum[bh*64 + d] = a0;
    ksum[(0*BHD + bh)*64 + d] = a1;
    ksum[(1*BHD + bh)*64 + d] = a2;
    ksum[(2*BHD + bh)*64 + d] = a3;
  }
}

// ---------------- pass B stats: per-(b,h) segment means -> 3-way softmax multipliers ----------------
__global__ void stats_kernel(const int* __restrict__ mask, const float* __restrict__ qsum,
                             const float* __restrict__ ksum, float* __restrict__ mult){
  __shared__ float cnt[BATCH][3];
  int t = threadIdx.x;
  if (t < BATCH*3){
    int b = t/3, s = t%3;
    int off = (s==0) ? 0 : (s==1 ? 256 : 768);
    int len = (s==1) ? 512 : 256;
    int c = 0;
    for (int i = 0; i < len; i++) c += (mask[b*NSEQ + off + i] != 0);
    cnt[b][s] = (float)c;
  }
  __syncthreads();
  if (t < BHD){
    int b = t / NHEAD;
    float m[3];
    for (int s = 0; s < 3; s++){
      float num = 0.f;
      for (int d = 0; d < 64; d++) num += qsum[t*64+d]*ksum[(s*BHD + t)*64 + d];
      num *= ATT_SCALE;
      float den = fmaxf(cnt[b][s]*1024.0f, 1.0f);
      float mm = num/den;
      m[s] = (mm == 0.0f) ? NEGINF : mm;
    }
    float mx = fmaxf(m[0], fmaxf(m[1], m[2]));
    if (mx == NEGINF){
      mult[0*BHD+t] = 0.f; mult[1*BHD+t] = 0.f; mult[2*BHD+t] = 0.f;
    } else {
      float e0 = __expf(m[0]-mx), e1 = __expf(m[1]-mx), e2 = __expf(m[2]-mx);
      float ss = e0+e1+e2;
      mult[0*BHD+t] = e0/ss; mult[1*BHD+t] = e1/ss; mult[2*BHD+t] = e2/ss;
    }
  }
}

// ---------------- segmented flash attention (S^T = K*Q^T so softmax is per-lane-column) ----------------
__global__ void attn_kernel(const f16* __restrict__ q, const f16* __restrict__ k,
                            const f16* __restrict__ vt, const float* __restrict__ mbias,
                            const float* __restrict__ mult, f16* __restrict__ out){
  int wid  = blockIdx.x*4 + (threadIdx.x >> 5);   // 6144 waves total
  int lane = threadIdx.x & 31;
  int qt = wid & 63;
  int h  = (wid >> 6) % NHEAD;
  int b  = wid / (64*NHEAD);
  int bh = b*NHEAD + h;
  const f16* qb = q  + ((size_t)bh*NSEQ + qt*16)*HDIM;
  const f16* kb = k  + (size_t)bh*NSEQ*HDIM;
  const f16* vb = vt + (size_t)bh*HDIM*NSEQ;
  const float* mb = mbias + b*NSEQ;

  // Q as B-fragment (columns), for d 0..31 and 32..63
  v16h bq0 = load_frag_b(qb,      HDIM, lane);
  v16h bq1 = load_frag_b(qb + 32, HDIM, lane);

  const v8f zv = {};
  v8f o[4]    = {};   // O^T segment accumulator: tile dt covers d rows dt*16..+15, col = query
  v8f oacc[4] = {};   // final accumulator across segments
  float mrun = NEGINF, lrun = 0.f;
  int hi = lane >> 4, col = lane & 15;
  int seg = 0;

  for (int kb0 = 0; kb0 < NSEQ; kb0 += 32){
    // S^T tiles: c0 keys kb0..+15, c1 keys kb0+16..+31; columns = 16 queries
    v8f c0 = zv, c1 = zv;
    v16h ak;
    ak = load_frag_a(kb + (size_t)(kb0     )*HDIM,      HDIM, lane); c0 = wmma32(ak, bq0, c0);
    ak = load_frag_a(kb + (size_t)(kb0     )*HDIM + 32, HDIM, lane); c0 = wmma32(ak, bq1, c0);
    ak = load_frag_a(kb + (size_t)(kb0 + 16)*HDIM,      HDIM, lane); c1 = wmma32(ak, bq0, c1);
    ak = load_frag_a(kb + (size_t)(kb0 + 16)*HDIM + 32, HDIM, lane); c1 = wmma32(ak, bq1, c1);

    // this lane's 16 key scores (C layout: lo lanes rows 0-7, hi lanes rows 8-15)
    int k0own = kb0 + 8*hi;
    int k1own = kb0 + 16 + 8*hi;
    float s0[8], s1[8];
    float sm = NEGINF;
    for (int r = 0; r < 8; r++){
      s0[r] = c0[r]*ATT_SCALE + mb[k0own + r];
      s1[r] = c1[r]*ATT_SCALE + mb[k1own + r];
      sm = fmaxf(sm, fmaxf(s0[r], s1[r]));
    }
    sm = fmaxf(sm, __shfl_xor(sm, 16, 32));       // sync column max across lane pair
    float mnew  = fmaxf(mrun, sm);
    float muse  = (mnew == NEGINF) ? 0.f : mnew;
    float alpha = (mrun == NEGINF) ? 0.f : __expf(mrun - mnew);

    float psum = 0.f;
    unsigned u0[4], u1[4];
    for (int j = 0; j < 4; j++){
      float p00 = __expf(s0[2*j]   - muse), p01 = __expf(s0[2*j+1] - muse);
      float p10 = __expf(s1[2*j]   - muse), p11 = __expf(s1[2*j+1] - muse);
      psum += p00 + p01 + p10 + p11;
      u0[j] = pack2h(p00, p01);
      u1[j] = pack2h(p10, p11);
    }
    lrun = lrun*alpha + psum;
    mrun = mnew;
    for (int dt = 0; dt < 4; dt++) o[dt] = o[dt]*alpha;

    // P^T C-layout -> B-fragment: only a half-wave exchange is needed
    unsigned cr[4];
    for (int j = 0; j < 4; j++) cr[j] = __shfl_xor(hi ? u0[j] : u1[j], 16, 32);
    union { v16h v; unsigned u[8]; } P;
    for (int j = 0; j < 4; j++){
      P.u[j]   = hi ? cr[j] : u0[j];
      P.u[4+j] = hi ? u1[j] : cr[j];
    }

    // O^T += V^T (A-frag, rows = d) x P^T (B-frag), K = 32 keys
    for (int dt = 0; dt < 4; dt++){
      v16h av = load_frag_a(vb + (size_t)(dt*16)*NSEQ + kb0, NSEQ, lane);
      o[dt] = wmma32(av, P.v, o[dt]);
    }

    // segment boundary? (boundaries 256/768/1024 are multiples of 32)
    int nxt = kb0 + 32;
    if (nxt == 256 || nxt == 768 || nxt == 1024){
      float ltot = lrun + __shfl_xor(lrun, 16, 32);
      float mv = mult[seg*BHD + bh];
      float factor = (ltot > 0.f && mv > 0.f) ? (mv/ltot) : 0.f;
      for (int dt = 0; dt < 4; dt++){ oacc[dt] += o[dt]*factor; o[dt] = zv; }
      mrun = NEGINF; lrun = 0.f;
      seg++;
    }
  }

  // store: out[b, q, h*64 + d] (f16), one 16B store per tile per lane
  int qg = qt*16 + col;
  for (int dt = 0; dt < 4; dt++){
    v8h hv;
    for (int r = 0; r < 8; r++) hv[r] = (f16)oacc[dt][r];
    int dbase = dt*16 + 8*hi;
    *(v8h*)(out + ((size_t)(b*NSEQ + qg))*DMODEL + h*64 + dbase) = hv;
  }
}

// ---------------- host ----------------
extern "C" void kernel_launch(void* const* d_in, const int* in_sizes, int n_in,
                              void* d_out, int out_size, void* d_ws, size_t ws_size,
                              hipStream_t stream){
  (void)in_sizes; (void)n_in; (void)out_size; (void)ws_size;
  const float* x     = (const float*)d_in[0];
  const int*   mask  = (const int*)  d_in[1];
  const float* ln1w  = (const float*)d_in[2];
  const float* ln1b  = (const float*)d_in[3];
  const float* qkvw[3] = {(const float*)d_in[4], (const float*)d_in[5], (const float*)d_in[6]};
  const float* projw = (const float*)d_in[7];
  const float* projb = (const float*)d_in[8];
  const float* ln2w  = (const float*)d_in[9];
  const float* ln2b  = (const float*)d_in[10];
  const float* fc1w  = (const float*)d_in[11];
  const float* fc1b  = (const float*)d_in[12];
  const float* fc2w  = (const float*)d_in[13];
  const float* fc2b  = (const float*)d_in[14];
  float* out = (float*)d_out;

  char* ws = (char*)d_ws;
  size_t off = 0;
  auto carve = [&](size_t bytes) -> void* {
    void* p = ws + off;
    off += (bytes + 255) & ~(size_t)255;
    return p;
  };
  const size_t QKV_W = (size_t)3*DMODEL*DMODEL;   // 2304*768 per segment
  f16* wh_qkv0 = (f16*)carve(QKV_W*2);
  f16* wh_qkv1 = (f16*)carve(QKV_W*2);
  f16* wh_qkv2 = (f16*)carve(QKV_W*2);
  f16* wh_proj = (f16*)carve((size_t)DMODEL*DMODEL*2);
  f16* wh_fc1  = (f16*)carve((size_t)4*DMODEL*DMODEL*2);
  f16* wh_fc2  = (f16*)carve((size_t)4*DMODEL*DMODEL*2);
  f16* bufA    = (f16*)carve((size_t)ROWS*DMODEL*2);            // xn / attn_out / h
  f16* region  = (f16*)carve((size_t)ROWS*4*DMODEL*2);          // q|k|vt, later ff1
  f16* qh  = region;
  f16* kh  = region + (size_t)ROWS*DMODEL;
  f16* vth = region + (size_t)2*ROWS*DMODEL;
  f16* ff1 = region;                                            // alias (q,k,v dead by then)
  float* x2    = (float*)carve((size_t)ROWS*DMODEL*4);
  float* qsum  = (float*)carve((size_t)BHD*64*4);
  float* ksum  = (float*)carve((size_t)3*BHD*64*4);
  float* multp = (float*)carve((size_t)3*BHD*4);
  float* mbias = (float*)carve((size_t)ROWS*4);

  // weight conversion f32 -> f16
  auto cvt = [&](const float* src, f16* dst, size_t n){
    cvt_f32_f16_kernel<<<dim3((unsigned)((n + 255)/256)), dim3(256), 0, stream>>>(src, dst, (int)n);
  };
  cvt(qkvw[0], wh_qkv0, QKV_W);
  cvt(qkvw[1], wh_qkv1, QKV_W);
  cvt(qkvw[2], wh_qkv2, QKV_W);
  cvt(projw, wh_proj, (size_t)DMODEL*DMODEL);
  cvt(fc1w,  wh_fc1,  (size_t)4*DMODEL*DMODEL);
  cvt(fc2w,  wh_fc2,  (size_t)4*DMODEL*DMODEL);

  maskbias_kernel<<<dim3(ROWS/256), dim3(256), 0, stream>>>(mask, mbias);

  // LN1
  ln_kernel<<<dim3(ROWS), dim3(256), 0, stream>>>(x, ln1w, ln1b, bufA);

  // QKV per segment: grid (Nout/128, seg/128, B), 128x128 per block
  const int seg_off_a[3] = {0, 256, 768};
  const int seg_len_a[3] = {256, 512, 256};
  f16* whs[3] = {wh_qkv0, wh_qkv1, wh_qkv2};
  for (int s = 0; s < 3; s++){
    gemm_kernel<0><<<dim3(18, seg_len_a[s]/128, BATCH), dim3(128), 0, stream>>>(
        bufA, whs[s], DMODEL, 3*DMODEL,
        nullptr, nullptr, nullptr, nullptr,
        qh, kh, vth, seg_off_a[s]);
  }

  // segment-mean stats
  sumqk_kernel<<<dim3(BHD), dim3(256), 0, stream>>>(qh, kh, mbias, qsum, ksum);
  stats_kernel<<<dim3(1), dim3(128), 0, stream>>>(mask, qsum, ksum, multp);

  // attention -> bufA (f16, (B,N,D))
  attn_kernel<<<dim3(BATCH*NHEAD*64/4), dim3(128), 0, stream>>>(qh, kh, vth, mbias, multp, bufA);

  // proj + bias + residual -> x2 (f32)
  gemm_kernel<1><<<dim3(DMODEL/128, ROWS/128, 1), dim3(128), 0, stream>>>(
      bufA, wh_proj, DMODEL, DMODEL, projb, x, x2, nullptr,
      nullptr, nullptr, nullptr, 0);

  // LN2 -> bufA (f16)
  ln_kernel<<<dim3(ROWS), dim3(256), 0, stream>>>(x2, ln2w, ln2b, bufA);

  // FC1 + bias + exact GELU -> ff1 (f16)
  gemm_kernel<2><<<dim3(4*DMODEL/128, ROWS/128, 1), dim3(128), 0, stream>>>(
      bufA, wh_fc1, DMODEL, 4*DMODEL, fc1b, nullptr, nullptr, ff1,
      nullptr, nullptr, nullptr, 0);

  // FC2 + bias + residual -> out (f32)
  gemm_kernel<1><<<dim3(DMODEL/128, ROWS/128, 1), dim3(128), 0, stream>>>(
      ff1, wh_fc2, 4*DMODEL, DMODEL, fc2b, x2, out, nullptr,
      nullptr, nullptr, nullptr, 0);
}
